// CrossAttention_85615878078502
// MI455X (gfx1250) — matrix operand order repaired and checked by reading, not compiled, not run on hardware
//
#include <hip/hip_runtime.h>

// ---------------------------------------------------------------------------
// CDNA5 (gfx1250) cross-attention.
//  - all GEMMs via v_wmma_f32_16x16x32_bf16 (wave32, ISA 7.12.2 layouts)
//  - one-time fp32->bf16 pre-conversion (native v_cvt, bf16 x/weights stay in L2)
//  - attention K head-slice staged by the Tensor Data Mover (tensor_load_to_lds
//    + s_wait_tensorcnt), with TDM row padding giving a conflict-free LDS stride
//  - global_prefetch of next GEMM tiles
// ---------------------------------------------------------------------------

typedef __attribute__((ext_vector_type(16))) __bf16 bf16x16;
typedef __attribute__((ext_vector_type(8)))  float  f32x8;
typedef __attribute__((ext_vector_type(4)))  unsigned int u32x4;
typedef __attribute__((ext_vector_type(8)))  int          i32x8;
typedef __attribute__((ext_vector_type(4)))  int          i32x4;

union FragAB {
  bf16x16      v;
  unsigned int u[8];
};

__device__ __forceinline__ unsigned short f32_to_bf16_bits(float f) {
  return __builtin_bit_cast(unsigned short, (__bf16)f);  // native cvt, RTNE
}

// ---------------------------------------------------------------------------
// one-time fp32 -> bf16 conversion (n is a multiple of 4 for all our tensors)
// ---------------------------------------------------------------------------
__global__ __launch_bounds__(256) void cvt_f32_bf16_kernel(
    const float* __restrict__ s, unsigned short* __restrict__ d, int n) {
  const int i = (blockIdx.x * 256 + threadIdx.x) * 4;
  if (i + 3 < n) {
    const float4 f = *(const float4*)(s + i);
    uint2 p;
    p.x = (unsigned)f32_to_bf16_bits(f.x) | ((unsigned)f32_to_bf16_bits(f.y) << 16);
    p.y = (unsigned)f32_to_bf16_bits(f.z) | ((unsigned)f32_to_bf16_bits(f.w) << 16);
    *(uint2*)(d + i) = p;
  } else {
    for (int j = i; j < n; ++j) d[j] = f32_to_bf16_bits(s[j]);
  }
}

// ---------------------------------------------------------------------------
// GEMM + bias:  C[M,N] = A[M,K] @ W[K,N] + bias[N]   (A, W: bf16 bits)
// Block tile 128x64, BK=32, 256 threads = 8 waves; wave w owns rows
// [w*16, w*16+16) x all 64 cols (4 x 16x16 f32 accumulators).
// ---------------------------------------------------------------------------
template <bool OUT_F32>
__global__ __launch_bounds__(256) void gemm_bias_wmma_kernel(
    const unsigned short* __restrict__ A, const unsigned short* __restrict__ W,
    const float* __restrict__ bias, void* __restrict__ Cptr,
    int M, int K, int N) {
  constexpr int BM = 128, BN = 64, BK = 32, LDT = 36;  // padded LDS stride
  __shared__ alignas(16) unsigned short As[BM * LDT];  // [m][k]
  __shared__ alignas(16) unsigned short Ws[BN * LDT];  // [n][k] (transposed)

  const int tid  = threadIdx.x;
  const int bm   = blockIdx.y * BM;
  const int bn   = blockIdx.x * BN;
  const int wave = tid >> 5;
  const int lane = tid & 31;
  const int l    = lane & 15;
  const int hi   = lane >> 4;
  const int m0l  = wave * 16;

  const f32x8 zero = {0.f, 0.f, 0.f, 0.f, 0.f, 0.f, 0.f, 0.f};
  f32x8 acc[4] = {zero, zero, zero, zero};

  for (int k0 = 0; k0 < K; k0 += BK) {
    // --- stage A tile (128x32 bf16) ---
#pragma unroll
    for (int p = 0; p < 4; ++p) {
      const int rl = (tid >> 3) + p * 32;  // 0..127
      const int cl = (tid & 7) * 4;        // 0..28
      const int grow = bm + rl;
      uint2 dv; dv.x = 0u; dv.y = 0u;
      if (grow < M) dv = *(const uint2*)(A + (size_t)grow * K + k0 + cl);
      *(uint2*)&As[rl * LDT + cl] = dv;    // (rl*36+cl)*2 is 8B aligned
    }
    // --- stage W tile (32x64 bf16) transposed -> Ws[n][k] ---
#pragma unroll
    for (int p = 0; p < 2; ++p) {
      const int kl = (tid >> 4) + p * 16;  // 0..31
      const int nl = (tid & 15) * 4;       // 0..60
      const uint2 wv = *(const uint2*)(W + (size_t)(k0 + kl) * N + bn + nl);
      Ws[(nl + 0) * LDT + kl] = (unsigned short)(wv.x & 0xFFFFu);
      Ws[(nl + 1) * LDT + kl] = (unsigned short)(wv.x >> 16);
      Ws[(nl + 2) * LDT + kl] = (unsigned short)(wv.y & 0xFFFFu);
      Ws[(nl + 3) * LDT + kl] = (unsigned short)(wv.y >> 16);
    }
    // prefetch next tiles into the cache hierarchy (global_prefetch_b8)
    if (k0 + BK < K) {
      const int rl = (tid >> 3);
      const int cl = (tid & 7) * 4;
      if (bm + rl < M)
        __builtin_prefetch(A + (size_t)(bm + rl) * K + k0 + BK + cl, 0, 1);
      __builtin_prefetch(W + (size_t)(k0 + BK + (tid >> 4)) * N + bn + (tid & 15) * 4, 0, 1);
    }
    __syncthreads();

    // A fragment (ISA 7.12.2): lane l = row, VGPR i -> K pairs
    FragAB af;
#pragma unroll
    for (int i = 0; i < 8; ++i) {
      const int koff = ((i < 4) ? 2 * i : 16 + 2 * (i - 4)) + hi * 8;
      af.u[i] = *(const unsigned int*)&As[(m0l + l) * LDT + koff];
    }
#pragma unroll
    for (int j = 0; j < 4; ++j) {
      FragAB bf;
#pragma unroll
      for (int i = 0; i < 8; ++i) {
        const int koff = ((i < 4) ? 2 * i : 16 + 2 * (i - 4)) + hi * 8;  // B: same K order as A fragment
        bf.u[i] = *(const unsigned int*)&Ws[(j * 16 + l) * LDT + koff];
      }
      acc[j] = __builtin_amdgcn_wmma_f32_16x16x32_bf16(
          false, af.v, false, bf.v, (short)0, acc[j], false, false);
    }
    __syncthreads();
  }

  // epilogue: C VGPR r -> row r + 8*hi, col = lane&15
#pragma unroll
  for (int j = 0; j < 4; ++j) {
    const int n = bn + j * 16 + l;
    const float bv = bias[n];
#pragma unroll
    for (int r = 0; r < 8; ++r) {
      const int row = bm + m0l + r + hi * 8;
      if (row < M) {
        const float val = acc[j][r] + bv;
        if (OUT_F32)
          ((float*)Cptr)[(size_t)row * N + n] = val;
        else
          ((unsigned short*)Cptr)[(size_t)row * N + n] = f32_to_bf16_bits(val);
      }
    }
  }
}

// ---------------------------------------------------------------------------
// Fused attention: one block = (batch b, head h, 64 q rows), 128 thr = 4 waves.
// K head-slice staged via TDM (tensor_load_to_lds) with LDS row padding:
// 64 elem rows + 1 DWORD pad -> stride 66 (33*n mod 64 is a permutation ->
// conflict-free B-fragment gathers). V staged manually (transposed).
// ---------------------------------------------------------------------------
__global__ __launch_bounds__(128) void attention_wmma_kernel(
    const unsigned short* __restrict__ Q, const unsigned short* __restrict__ Kb,
    const unsigned short* __restrict__ Vb, unsigned short* __restrict__ O) {
  constexpr int SKV = 77, DH = 64, SQ = 4096, NE = 1024;
  constexpr int KP = 80, KP2 = 96;
  constexpr int LDK = 66, LDV = 100, LDP = 100;

  // single LDS block so the K tile sits at LDS byte offset 0 (TDM lds_addr)
  __shared__ alignas(16) unsigned short smem[KP * LDK + DH * LDV + 4 * 16 * LDP];
  unsigned short* Ks  = smem;                       // [key][dim], stride 66
  unsigned short* VTs = smem + KP * LDK;            // [dim][key] (transposed)
  unsigned short* Ps  = smem + KP * LDK + DH * LDV; // per-wave P [16][key]

  const int tid  = threadIdx.x;
  const int qblk = blockIdx.x & 63;
  const int bh   = blockIdx.x >> 6;
  const int h    = bh & 15;
  const int b    = bh >> 4;
  const int wave = tid >> 5, lane = tid & 31;
  const int l = lane & 15, hi = lane >> 4;

#if __has_builtin(__builtin_amdgcn_tensor_load_to_lds) && __has_builtin(__builtin_amdgcn_s_wait_tensorcnt)
  #define ATTN_USE_TDM 1
#else
  #define ATTN_USE_TDM 0
#endif

#if ATTN_USE_TDM
  if (wave == 0) {
    // D# for a 2D tile: 64 (contig dims) x 77 (keys), element 2B, row stride 1024,
    // pad 1 DWORD after every 32 DWORDs (=one 64-elem row) -> LDS stride 66.
    const unsigned long long ga =
        (unsigned long long)(const void*)(Kb + (size_t)(b * SKV) * NE + h * DH);
    u32x4 g0;
    g0[0] = 1u;                                           // count=1, user D#
    g0[1] = 0u;                                           // lds_addr = 0 (Ks base)
    g0[2] = (unsigned int)(ga & 0xFFFFFFFFu);             // global_addr lo
    g0[3] = (unsigned int)((ga >> 32) & 0x01FFFFFFu) | 0x80000000u;  // hi | type=2
    i32x8 g1;
    g1[0] = (int)((1u << 16) | (1u << 20) | (4u << 22));  // data_size=2B, pad_en, pad_interval=32DW, pad_amount=1DW
    g1[1] = (int)(1024u << 16);                           // tensor_dim0 = 1024 (lo16)
    g1[2] = (int)((unsigned)SKV << 16);                   // tensor_dim0 hi=0, tensor_dim1 lo=77
    g1[3] = (int)(64u << 16);                             // tensor_dim1 hi=0, tile_dim0=64
    g1[4] = SKV;                                          // tile_dim1=77, tile_dim2=0
    g1[5] = 1024;                                         // tensor_dim0_stride lo
    g1[6] = 0;                                            // stride hi / dim1_stride lo
    g1[7] = 0;
    i32x4 gz4 = {0, 0, 0, 0};
    i32x8 gz8 = {0, 0, 0, 0, 0, 0, 0, 0};
    // 6-arg form (clang-23 / amdgpu-toolchain): extra zero-filled i32x8 group.
    __builtin_amdgcn_tensor_load_to_lds(g0, g1, gz4, gz4, gz8, 0);
  }
#else
  for (int idx = tid; idx < SKV * DH; idx += 128) {
    const int key = idx / DH, d = idx % DH;
    Ks[key * LDK + d] = Kb[(size_t)(b * SKV + key) * NE + h * DH + d];
  }
#endif

  // stage V transposed (overlaps with the TDM DMA), zero-pad keys 77..95
  for (int idx = tid; idx < KP2 * DH; idx += 128) {
    const int key = idx / 64, d = idx & 63;
    unsigned short v = 0;
    if (key < SKV) v = Vb[(size_t)(b * SKV + key) * NE + h * DH + d];
    VTs[d * LDV + key] = v;
  }
  // zero-pad K rows 77..79 (TDM writes only 77 rows)
  for (int idx = tid; idx < (KP - SKV) * DH; idx += 128) {
    const int key = SKV + idx / DH, d = idx % DH;
    Ks[key * LDK + d] = 0;
  }
#if ATTN_USE_TDM
  if (wave == 0) __builtin_amdgcn_s_wait_tensorcnt(0);
#endif
  __syncthreads();

  const int qrow = b * SQ + qblk * 64 + wave * 16;  // global q-row base
  const f32x8 zero = {0.f, 0.f, 0.f, 0.f, 0.f, 0.f, 0.f, 0.f};

  // Q fragments for the two k-steps of D_HEAD=64
  FragAB aq[2];
  const unsigned int* Q32 = (const unsigned int*)Q;
#pragma unroll
  for (int s = 0; s < 2; ++s) {
#pragma unroll
    for (int i = 0; i < 8; ++i) {
      const int koff = ((i < 4) ? 2 * i : 16 + 2 * (i - 4)) + hi * 8;
      aq[s].u[i] = Q32[((size_t)(qrow + l) * NE + h * DH + s * 32 + koff) >> 1];
    }
  }

  // scores: 5 key-chunks x 2 k-steps = 10 WMMAs
  f32x8 sf[5];
#pragma unroll
  for (int c = 0; c < 5; ++c) {
    f32x8 a = zero;
#pragma unroll
    for (int s = 0; s < 2; ++s) {
      FragAB bf;
#pragma unroll
      for (int i = 0; i < 8; ++i) {
        const int koff = s * 32 + ((i < 4) ? 2 * i : 16 + 2 * (i - 4)) + hi * 8;
        bf.u[i] = *(const unsigned int*)&Ks[(c * 16 + l) * LDK + koff];
      }
      a = __builtin_amdgcn_wmma_f32_16x16x32_bf16(
          false, aq[s].v, false, bf.v, (short)0, a, false, false);
    }
    sf[c] = a;
  }

  // masked row softmax; C-layout: row = r + 8*hi, key = c*16 + (lane&15)
  const float scale = 0.125f;  // 1/sqrt(64)
  float rinv[8];
#pragma unroll
  for (int r = 0; r < 8; ++r) {
    float sv[5], m = -3.0e38f;
#pragma unroll
    for (int c = 0; c < 5; ++c) {
      const int key = c * 16 + l;
      const float x = sf[c][r] * scale;
      sv[c] = (key < SKV) ? x : -3.0e38f;
      m = fmaxf(m, sv[c]);
    }
#pragma unroll
    for (int off = 8; off >= 1; off >>= 1) m = fmaxf(m, __shfl_xor(m, off, 16));
    float sum = 0.f;
    const int prow = wave * 16 + r + hi * 8;
#pragma unroll
    for (int c = 0; c < 5; ++c) {
      const int key = c * 16 + l;
      const float p = (key < SKV) ? __expf(sv[c] - m) : 0.f;
      sum += p;
      Ps[prow * LDP + key] = f32_to_bf16_bits(p);
    }
    Ps[prow * LDP + 80 + l] = 0;  // zero-pad keys 80..95
#pragma unroll
    for (int off = 8; off >= 1; off >>= 1) sum += __shfl_xor(sum, off, 16);
    rinv[r] = 1.f / sum;
  }
  __syncthreads();

  // O = P @ V : 3 key-steps x 4 dim-chunks = 12 WMMAs
  f32x8 of[4] = {zero, zero, zero, zero};
#pragma unroll
  for (int s = 0; s < 3; ++s) {
    FragAB ap;
#pragma unroll
    for (int i = 0; i < 8; ++i) {
      const int koff = s * 32 + ((i < 4) ? 2 * i : 16 + 2 * (i - 4)) + hi * 8;
      ap.u[i] = *(const unsigned int*)&Ps[(wave * 16 + l) * LDP + koff];
    }
#pragma unroll
    for (int ch = 0; ch < 4; ++ch) {
      FragAB bf;
#pragma unroll
      for (int i = 0; i < 8; ++i) {
        const int koff = s * 32 + ((i < 4) ? 2 * i : 16 + 2 * (i - 4)) + hi * 8;
        bf.u[i] = *(const unsigned int*)&VTs[(ch * 16 + l) * LDV + koff];
      }
      of[ch] = __builtin_amdgcn_wmma_f32_16x16x32_bf16(
          false, ap.v, false, bf.v, (short)0, of[ch], false, false);
    }
  }

  // normalize + store bf16 attn output
#pragma unroll
  for (int ch = 0; ch < 4; ++ch) {
#pragma unroll
    for (int r = 0; r < 8; ++r) {
      const float val = of[ch][r] * rinv[r];
      const int row = qrow + r + hi * 8;
      O[(size_t)row * NE + h * DH + ch * 16 + l] = f32_to_bf16_bits(val);
    }
  }
}

// ---------------------------------------------------------------------------
extern "C" void kernel_launch(void* const* d_in, const int* in_sizes, int n_in,
                              void* d_out, int out_size, void* d_ws, size_t ws_size,
                              hipStream_t stream) {
  (void)in_sizes; (void)n_in; (void)out_size; (void)ws_size;
  const float* x  = (const float*)d_in[0];
  const float* y  = (const float*)d_in[1];
  const float* wq = (const float*)d_in[2];
  const float* bq = (const float*)d_in[3];
  const float* wk = (const float*)d_in[4];
  const float* bk = (const float*)d_in[5];
  const float* wv = (const float*)d_in[6];
  const float* bv = (const float*)d_in[7];
  const float* wo = (const float*)d_in[8];
  const float* bo = (const float*)d_in[9];
  float* out = (float*)d_out;

  const int B = 8, SQ = 4096, SKV = 77, DE = 1024, DC = 768, NH = 16;
  const int MQ = B * SQ;     // 32768
  const int MKV = B * SKV;   // 616

  // ---- workspace layout (bf16 bit buffers) ----
  unsigned short* x_bf  = (unsigned short*)d_ws;               // MQ  x DE
  unsigned short* y_bf  = x_bf  + (size_t)MQ * DE;             // MKV x DC
  unsigned short* wq_bf = y_bf  + (size_t)MKV * DC;            // DE x DE
  unsigned short* wk_bf = wq_bf + (size_t)DE * DE;             // DC x DE
  unsigned short* wv_bf = wk_bf + (size_t)DC * DE;             // DC x DE
  unsigned short* wo_bf = wv_bf + (size_t)DC * DE;             // DE x DE
  unsigned short* q_bf  = wo_bf + (size_t)DE * DE;             // MQ  x DE
  unsigned short* k_bf  = q_bf  + (size_t)MQ * DE;             // MKV x DE
  unsigned short* v_bf  = k_bf  + (size_t)MKV * DE;            // MKV x DE
  unsigned short* a_bf  = x_bf;  // alias: x_bf dead after Q projection

  const dim3 blk(256);
  auto cvt = [&](const float* s, unsigned short* d, int n) {
    cvt_f32_bf16_kernel<<<dim3((n / 4 + 255) / 256), blk, 0, stream>>>(s, d, n);
  };
  // one-time bf16 conversions (bf16 x + weights fit in 192MB L2)
  cvt(x,  x_bf,  MQ * DE);
  cvt(y,  y_bf,  MKV * DC);
  cvt(wq, wq_bf, DE * DE);
  cvt(wk, wk_bf, DC * DE);
  cvt(wv, wv_bf, DC * DE);
  cvt(wo, wo_bf, DE * DE);

  // Q = x @ wq + bq  -> bf16
  gemm_bias_wmma_kernel<false><<<dim3(DE / 64, MQ / 128), blk, 0, stream>>>(
      x_bf, wq_bf, bq, q_bf, MQ, DE, DE);
  // K = y @ wk + bk  -> bf16
  gemm_bias_wmma_kernel<false><<<dim3(DE / 64, (MKV + 127) / 128), blk, 0, stream>>>(
      y_bf, wk_bf, bk, k_bf, MKV, DC, DE);
  // V = y @ wv + bv  -> bf16
  gemm_bias_wmma_kernel<false><<<dim3(DE / 64, (MKV + 127) / 128), blk, 0, stream>>>(
      y_bf, wv_bf, bv, v_bf, MKV, DC, DE);
  // fused softmax(QK^T/sqrt(d)) @ V  -> bf16
  attention_wmma_kernel<<<dim3(B * NH * (SQ / 64)), dim3(128), 0, stream>>>(
      q_bf, k_bf, v_bf, a_bf);
  // out = attn @ wo + bo -> fp32
  gemm_bias_wmma_kernel<true><<<dim3(DE / 64, MQ / 128), blk, 0, stream>>>(
      a_bf, wo_bf, bo, out, MQ, DE, DE);
}
